// TensorFusionModel_15685220565459
// MI455X (gfx1250) — matrix operand order repaired and checked by reading, not compiled, not run on hardware
//
#include <hip/hip_runtime.h>
#include <hip/hip_bf16.h>
#include <math.h>

// ---------------------------------------------------------------------------
// CDNA5 (gfx1250, wave32) implementation of the TensorFusion model.
// Dense GEMMs use v_wmma_f32_16x16x32_f16 (fp16 in, fp32 accumulate).
// LDS is laid out so every WMMA fragment is two contiguous 16-byte runs
// (ds_load_b128), per the ISA fragment tables:
//   A 16x32 frag of lane(h,r): As[row=r][h*8 .. +8) ++ As[row][16+h*8 .. +8)
//   B 32x16 frag of lane(h,r): Bst[col=r][h*8 .. +8) ++ Bst[col][16+h*8 .. +8)   (B stored transposed)
// ---------------------------------------------------------------------------

typedef __attribute__((ext_vector_type(16))) _Float16 v16h;
typedef __attribute__((ext_vector_type(8)))  _Float16 v8h;
typedef __attribute__((ext_vector_type(4)))  _Float16 v4h;
typedef __attribute__((ext_vector_type(8)))  float    v8f;

#define HID   64
#define HEADS 4
#define DH    16
#define BATCH 128

static __device__ __forceinline__ v16h cat8(v8h lo, v8h hi) {
  return __builtin_shufflevector(lo, hi, 0, 1, 2, 3, 4, 5, 6, 7,
                                 8, 9, 10, 11, 12, 13, 14, 15);
}

// ---------------------------------------------------------------------------
// Generic tiled GEMM:  C[M,N] = epi( A[M,K] @ W[K,N] + bias [+resid] )
// Block = 128 threads (4 waves), tile 64(M) x 64(N), K step 64.
// Requires M%64==0, N%64==0, K%64==0 (true for every encoder GEMM here).
// EPI: 0 = bias, 1 = bias + sinusoidal pos-enc, 2 = bias + exact GELU,
//      3 = bias + residual (pre-LN sum).
// ---------------------------------------------------------------------------
template <int EPI>
__global__ __launch_bounds__(128)
void gemm64_wmma(const float* __restrict__ A, const float* __restrict__ W,
                 const float* __restrict__ bias, const float* __restrict__ resid,
                 float* __restrict__ C, int M, int N, int K, int T) {
  __shared__ __align__(16) _Float16 As [64][72];   // [m][k], stride 144B
  __shared__ __align__(16) _Float16 Bst[64][72];   // [n][k] (transposed)

  const int tid  = threadIdx.x;
  const int bm   = blockIdx.y;
  const int bn   = blockIdx.x;
  const int wave = tid >> 5;
  const int lane = tid & 31;
  const int half_sel = lane >> 4;
  const int r        = lane & 15;
  const int wm = wave >> 1;          // 0..1 -> 32-row band
  const int wn = wave & 1;           // 0..1 -> 32-col band

  v8f acc[2][2] = {};

  for (int k0 = 0; k0 < K; k0 += 64) {
    // --- stage A tile (64 x 64) fp32 -> fp16, vectorized ---
#pragma unroll
    for (int it = 0; it < 8; ++it) {
      int idx4 = it * 128 + tid;            // 1024 float4 quads
      int row = idx4 >> 4, q = idx4 & 15;
      float4 f = *(const float4*)&A[(size_t)(bm * 64 + row) * K + k0 + q * 4];
      v4h hq = { (_Float16)f.x, (_Float16)f.y, (_Float16)f.z, (_Float16)f.w };
      *(v4h*)&As[row][q * 4] = hq;
    }
    // --- stage B tile (64k x 64n), transposed into Bst[n][k] ---
#pragma unroll
    for (int it = 0; it < 8; ++it) {
      int idx4 = it * 128 + tid;
      int krow = idx4 >> 4, q = idx4 & 15;
      float4 f = *(const float4*)&W[(size_t)(k0 + krow) * N + bn * 64 + q * 4];
      Bst[q * 4 + 0][krow] = (_Float16)f.x;
      Bst[q * 4 + 1][krow] = (_Float16)f.y;
      Bst[q * 4 + 2][krow] = (_Float16)f.z;
      Bst[q * 4 + 3][krow] = (_Float16)f.w;
    }
    // prefetch next K tile (global_prefetch_b8)
    if (k0 + 64 < K) {
      __builtin_prefetch(&A[(size_t)(bm * 64 + (tid >> 1)) * K + k0 + 64 + (tid & 1) * 32], 0, 1);
      __builtin_prefetch(&W[(size_t)(k0 + 64 + (tid >> 1)) * N + bn * 64 + (tid & 1) * 32], 0, 1);
    }
    __syncthreads();

#pragma unroll
    for (int ks = 0; ks < 64; ks += 32) {
      v16h bfr[2];
#pragma unroll
      for (int tn = 0; tn < 2; ++tn) {
        int bcol = wn * 32 + tn * 16 + r;
        v8h blo = *(const v8h*)&Bst[bcol][ks + half_sel * 8];
        v8h bhi = *(const v8h*)&Bst[bcol][ks + 16 + half_sel * 8];
        bfr[tn] = cat8(blo, bhi);
      }
#pragma unroll
      for (int tm = 0; tm < 2; ++tm) {
        int arow = wm * 32 + tm * 16 + r;
        v8h alo = *(const v8h*)&As[arow][ks + half_sel * 8];
        v8h ahi = *(const v8h*)&As[arow][ks + 16 + half_sel * 8];
        v16h a = cat8(alo, ahi);
#pragma unroll
        for (int tn = 0; tn < 2; ++tn)
          acc[tm][tn] = __builtin_amdgcn_wmma_f32_16x16x32_f16(
              false, a, false, bfr[tn], (short)0, acc[tm][tn], false, false);
      }
    }
    __syncthreads();
  }

  // --- epilogue (EPI compiled in) ---
#pragma unroll
  for (int tm = 0; tm < 2; ++tm) {
#pragma unroll
    for (int tn = 0; tn < 2; ++tn) {
#pragma unroll
      for (int i8 = 0; i8 < 8; ++i8) {
        int grow = bm * 64 + wm * 32 + tm * 16 + half_sel * 8 + i8;
        int gcol = bn * 64 + wn * 32 + tn * 16 + r;
        float v = acc[tm][tn][i8] + bias[gcol];
        if (EPI == 1) {               // + positional encoding (col dim = 64)
          int t = grow % T;
          int i2 = (gcol >> 1) << 1;
          float div = __expf(-(float)i2 * (9.210340371976184f / 64.0f));
          float ang = (float)t * div;
          v += (gcol & 1) ? __cosf(ang) : __sinf(ang);
        } else if (EPI == 2) {        // exact GELU
          v = 0.5f * v * (1.0f + erff(v * 0.7071067811865475f));
        } else if (EPI == 3) {        // residual add (pre-LN)
          v += resid[(size_t)grow * N + gcol];
        }
        C[(size_t)grow * N + gcol] = v;
      }
    }
  }
}

// ---------------------------------------------------------------------------
// Attention: one block per (batch, head); K/V staged in LDS, online softmax.
// qkv layout [B, T, 192]: q = cols 0..63, k = 64..127, v = 128..191.
// ---------------------------------------------------------------------------
__global__ __launch_bounds__(256)
void attn_kernel(const float* __restrict__ qkv, const int* __restrict__ mask,
                 float* __restrict__ o, int T) {
  __shared__ float Ks[256 * DH];
  __shared__ float Vs[256 * DH];
  __shared__ float bias_s[256];

  const int bh = blockIdx.x;
  const int b  = bh / HEADS;
  const int h  = bh % HEADS;
  const int tid = threadIdx.x;
  const float* base = qkv + (size_t)b * T * 192;

  for (int t = tid; t < T; t += blockDim.x) {
#pragma unroll
    for (int c = 0; c < DH; c += 4) {
      float4 fk = *(const float4*)&base[t * 192 + 64  + h * DH + c];
      float4 fv = *(const float4*)&base[t * 192 + 128 + h * DH + c];
      *(float4*)&Ks[t * DH + c] = fk;
      *(float4*)&Vs[t * DH + c] = fv;
    }
    bias_s[t] = (mask[(size_t)b * T + t] > 0) ? 0.0f : -1e9f;
  }
  __syncthreads();

  for (int t = tid; t < T; t += blockDim.x) {
    float q[DH];
#pragma unroll
    for (int c = 0; c < DH; ++c) q[c] = base[t * 192 + h * DH + c];
    float m = -1e30f, l = 0.0f, acc[DH];
#pragma unroll
    for (int c = 0; c < DH; ++c) acc[c] = 0.0f;
    for (int j = 0; j < T; ++j) {
      float s = 0.0f;
#pragma unroll
      for (int c = 0; c < DH; ++c) s += q[c] * Ks[j * DH + c];
      s = s * 0.25f + bias_s[j];            // 1/sqrt(16)
      float mn = fmaxf(m, s);
      float corr = __expf(m - mn);
      float p = __expf(s - mn);
      l = l * corr + p;
#pragma unroll
      for (int c = 0; c < DH; ++c) acc[c] = acc[c] * corr + p * Vs[j * DH + c];
      m = mn;
    }
    float invl = 1.0f / fmaxf(l, 1e-20f);
#pragma unroll
    for (int c = 0; c < DH; ++c)
      o[((size_t)b * T + t) * HID + h * DH + c] = acc[c] * invl;
  }
}

// ---------------------------------------------------------------------------
// LayerNorm over last dim (64), one wave per row (wave32: 2 cols/lane).
// ---------------------------------------------------------------------------
__global__ __launch_bounds__(32)
void ln64_kernel(float* __restrict__ x, const float* __restrict__ g,
                 const float* __restrict__ b) {
  const size_t row = blockIdx.x;
  const int lane = threadIdx.x;
  float x0 = x[row * 64 + lane];
  float x1 = x[row * 64 + lane + 32];
  float s = x0 + x1;
#pragma unroll
  for (int o = 16; o > 0; o >>= 1) s += __shfl_xor(s, o, 32);
  float mu = s * (1.0f / 64.0f);
  float d0 = x0 - mu, d1 = x1 - mu;
  float v = d0 * d0 + d1 * d1;
#pragma unroll
  for (int o = 16; o > 0; o >>= 1) v += __shfl_xor(v, o, 32);
  float inv = rsqrtf(v * (1.0f / 64.0f) + 1e-5f);
  x[row * 64 + lane]      = d0 * inv * g[lane]      + b[lane];
  x[row * 64 + lane + 32] = d1 * inv * g[lane + 32] + b[lane + 32];
}

// ---------------------------------------------------------------------------
// Masked mean pool -> z_[b, 0..64] with leading 1 (the "ones" concat).
// ---------------------------------------------------------------------------
__global__ __launch_bounds__(64)
void pool_kernel(const float* __restrict__ h, const int* __restrict__ mask,
                 float* __restrict__ z, int T) {
  const int b = blockIdx.x, d = threadIdx.x;
  float s = 0.0f, ms = 0.0f;
  for (int t = 0; t < T; ++t) {
    float mm = (mask[(size_t)b * T + t] > 0) ? 1.0f : 0.0f;
    s  += h[((size_t)b * T + t) * HID + d] * mm;
    ms += mm;
  }
  z[b * 65 + 1 + d] = s / fmaxf(ms, 1e-6f);
  if (d == 0) z[b * 65] = 1.0f;
}

// hh[b,n] = b1[n]   (split-K accumulation base)
__global__ void init_hh_kernel(float* __restrict__ hh, const float* __restrict__ b1) {
  int idx = blockIdx.x * blockDim.x + threadIdx.x;  // 16384
  hh[idx] = b1[idx & 127];
}

// ---------------------------------------------------------------------------
// Fusion GEMM: hh[b,n] += sum_{j,k} zt[b,i]*za[b,j]*zv[b,k] * w1[(i*65+j)*65+k, n]
// One block per i (65 blocks). 256 threads = 8 waves; 128x128 output tile,
// each wave owns 32x64 (2x4 WMMA tiles). A tile generated on the fly; K
// padded with zeros up to a multiple of 64.
// ---------------------------------------------------------------------------
__global__ __launch_bounds__(256)
void fusion_gemm_wmma(const float* __restrict__ zt, const float* __restrict__ za,
                      const float* __restrict__ zv, const float* __restrict__ w1,
                      float* __restrict__ hh) {
  __shared__ __align__(16) _Float16 As [128][72];   // [b][k]
  __shared__ __align__(16) _Float16 Bst[128][72];   // [n][k] (transposed)

  const int i   = blockIdx.x;        // 0..64
  const int tid = threadIdx.x;
  const int wave = tid >> 5;
  const int lane = tid & 31;
  const int half_sel = lane >> 4;
  const int r        = lane & 15;
  const int wm = wave >> 1;          // 0..3 -> 32-row bands
  const int wn = wave & 1;           // 0..1 -> 64-col bands
  const int KC = 65 * 65;            // 4225

  v8f acc[2][4] = {};

  for (int k0 = 0; k0 < KC; k0 += 64) {
    // --- generate A tile (128 x 64) on the fly ---
#pragma unroll
    for (int it = 0; it < 8; ++it) {
      int idx4 = it * 256 + tid;            // 2048 quads
      int row = idx4 >> 4, q = idx4 & 15;
      float zti = zt[row * 65 + i];
      v4h hq;
#pragma unroll
      for (int c = 0; c < 4; ++c) {
        int jk = k0 + q * 4 + c;
        float v = 0.0f;
        if (jk < KC) {
          int j = jk / 65, k = jk - j * 65;
          v = zti * za[row * 65 + j] * zv[row * 65 + k];
        }
        hq[c] = (_Float16)v;
      }
      *(v4h*)&As[row][q * 4] = hq;
    }
    // --- stage B tile (64k x 128n) from w1, transposed ---
#pragma unroll
    for (int it = 0; it < 8; ++it) {
      int idx4 = it * 256 + tid;            // 2048 quads
      int krow = idx4 >> 5, q = idx4 & 31;
      int jk = k0 + krow;
      float4 f = { 0.f, 0.f, 0.f, 0.f };
      if (jk < KC) f = *(const float4*)&w1[((size_t)i * KC + jk) * 128 + q * 4];
      Bst[q * 4 + 0][krow] = (_Float16)f.x;
      Bst[q * 4 + 1][krow] = (_Float16)f.y;
      Bst[q * 4 + 2][krow] = (_Float16)f.z;
      Bst[q * 4 + 3][krow] = (_Float16)f.w;
    }
    if (k0 + 64 < KC)
      __builtin_prefetch(&w1[((size_t)i * KC + k0 + 64 + (tid >> 2)) * 128 + (tid & 3) * 32], 0, 1);
    __syncthreads();

#pragma unroll
    for (int ks = 0; ks < 64; ks += 32) {
      v16h bfr[4];
#pragma unroll
      for (int tn = 0; tn < 4; ++tn) {
        int bcol = wn * 64 + tn * 16 + r;
        v8h blo = *(const v8h*)&Bst[bcol][ks + half_sel * 8];
        v8h bhi = *(const v8h*)&Bst[bcol][ks + 16 + half_sel * 8];
        bfr[tn] = cat8(blo, bhi);
      }
#pragma unroll
      for (int tm = 0; tm < 2; ++tm) {
        int arow = wm * 32 + tm * 16 + r;
        v8h alo = *(const v8h*)&As[arow][ks + half_sel * 8];
        v8h ahi = *(const v8h*)&As[arow][ks + 16 + half_sel * 8];
        v16h a = cat8(alo, ahi);
#pragma unroll
        for (int tn = 0; tn < 4; ++tn)
          acc[tm][tn] = __builtin_amdgcn_wmma_f32_16x16x32_f16(
              false, a, false, bfr[tn], (short)0, acc[tm][tn], false, false);
      }
    }
    __syncthreads();
  }

#pragma unroll
  for (int tm = 0; tm < 2; ++tm)
#pragma unroll
    for (int tn = 0; tn < 4; ++tn)
#pragma unroll
      for (int i8 = 0; i8 < 8; ++i8) {
        int row = wm * 32 + tm * 16 + half_sel * 8 + i8;
        int col = wn * 64 + tn * 16 + r;
        atomicAdd(&hh[row * 128 + col], acc[tm][tn][i8]);
      }
}

// out[b] = sum_n relu(hh[b,n]) * w2[n] + b2
__global__ __launch_bounds__(128)
void final_kernel(const float* __restrict__ hh, const float* __restrict__ w2,
                  const float* __restrict__ b2, float* __restrict__ out) {
  __shared__ float red[4];
  const int b = blockIdx.x, tid = threadIdx.x;
  float v = fmaxf(hh[b * 128 + tid], 0.0f) * w2[tid];
#pragma unroll
  for (int o = 16; o > 0; o >>= 1) v += __shfl_xor(v, o, 32);
  if ((tid & 31) == 0) red[tid >> 5] = v;
  __syncthreads();
  if (tid == 0) out[b] = red[0] + red[1] + red[2] + red[3] + b2[0];
}

// ---------------------------------------------------------------------------
// Host orchestration
// ---------------------------------------------------------------------------
extern "C" void kernel_launch(void* const* d_in, const int* in_sizes, int n_in,
                              void* d_out, int out_size, void* d_ws, size_t ws_size,
                              hipStream_t stream) {
  (void)in_sizes; (void)n_in; (void)out_size; (void)ws_size;

  // Workspace layout (floats). Peak use ~76 MB; modalities reuse buffers.
  float* ws   = (float*)d_ws;
  float* hbuf = ws;                         // 32768*64  = 2,097,152
  float* rbuf = ws + 2097152;               // 32768*64
  float* qkvb = ws + 4194304;               // 32768*192 = 6,291,456
  float* ffb  = ws + 10485760;              // 32768*256 = 8,388,608
  float* zb0  = ws + 18874368;              // 128*65
  float* zb1  = zb0 + 8320;
  float* zb2  = zb1 + 8320;
  float* hh   = zb2 + 8320;                 // 128*128
  float* zbuf[3] = { zb0, zb1, zb2 };

  // Inputs in setup_inputs() dict insertion order.
  struct Mod { const float* x; const int* mask; int T; int Din; int pbase; };
  Mod mods[3] = {
    { (const float*)d_in[0], (const int*)d_in[3],  64, 768,  6 },  // text
    { (const float*)d_in[1], (const int*)d_in[4], 256, 128, 20 },  // audio
    { (const float*)d_in[2], (const int*)d_in[5], 128, 512, 34 },  // vision
  };

  for (int m = 0; m < 3; ++m) {
    const int pb = mods[m].pbase;
    const float* proj_w = (const float*)d_in[pb + 0];
    const float* proj_b = (const float*)d_in[pb + 1];
    const float* qkv_w  = (const float*)d_in[pb + 2];
    const float* qkv_b  = (const float*)d_in[pb + 3];
    const float* out_w  = (const float*)d_in[pb + 4];
    const float* out_b  = (const float*)d_in[pb + 5];
    const float* ln1_g  = (const float*)d_in[pb + 6];
    const float* ln1_b  = (const float*)d_in[pb + 7];
    const float* ff1_w  = (const float*)d_in[pb + 8];
    const float* ff1_b  = (const float*)d_in[pb + 9];
    const float* ff2_w  = (const float*)d_in[pb + 10];
    const float* ff2_b  = (const float*)d_in[pb + 11];
    const float* ln2_g  = (const float*)d_in[pb + 12];
    const float* ln2_b  = (const float*)d_in[pb + 13];

    const int T = mods[m].T, M = BATCH * T, Din = mods[m].Din;

    // input projection + pos-enc
    gemm64_wmma<1><<<dim3(1, M / 64), 128, 0, stream>>>(
        mods[m].x, proj_w, proj_b, nullptr, hbuf, M, 64, Din, T);

    for (int l = 0; l < 2; ++l) {
      gemm64_wmma<0><<<dim3(3, M / 64), 128, 0, stream>>>(
          hbuf, qkv_w + (size_t)l * 64 * 192, qkv_b + l * 192, nullptr,
          qkvb, M, 192, 64, T);
      attn_kernel<<<BATCH * HEADS, 256, 0, stream>>>(qkvb, mods[m].mask, rbuf, T);
      gemm64_wmma<3><<<dim3(1, M / 64), 128, 0, stream>>>(
          rbuf, out_w + (size_t)l * 64 * 64, out_b + l * 64, hbuf,
          hbuf, M, 64, 64, T);
      ln64_kernel<<<M, 32, 0, stream>>>(hbuf, ln1_g + l * 64, ln1_b + l * 64);
      gemm64_wmma<2><<<dim3(4, M / 64), 128, 0, stream>>>(
          hbuf, ff1_w + (size_t)l * 64 * 256, ff1_b + l * 256, nullptr,
          ffb, M, 256, 64, T);
      gemm64_wmma<3><<<dim3(1, M / 64), 128, 0, stream>>>(
          ffb, ff2_w + (size_t)l * 256 * 64, ff2_b + l * 64, hbuf,
          hbuf, M, 64, 256, T);
      ln64_kernel<<<M, 32, 0, stream>>>(hbuf, ln2_g + l * 64, ln2_b + l * 64);
    }
    pool_kernel<<<BATCH, 64, 0, stream>>>(hbuf, mods[m].mask, zbuf[m], T);
  }

  // Tensor-fusion regression head
  const float* w1 = (const float*)d_in[48];
  const float* b1 = (const float*)d_in[49];
  const float* w2 = (const float*)d_in[50];
  const float* b2 = (const float*)d_in[51];

  init_hh_kernel<<<64, 256, 0, stream>>>(hh, b1);
  fusion_gemm_wmma<<<65, 256, 0, stream>>>(zb0, zb1, zb2, w1, hh);
  final_kernel<<<BATCH, 128, 0, stream>>>(hh, w2, b2, (float*)d_out);
}
